// MuiltiHeadAttention_90005334655603
// MI455X (gfx1250) — matrix operand order repaired and checked by reading, not compiled, not run on hardware
//
#include <hip/hip_runtime.h>
#include <hip/hip_bf16.h>

// ---------------------------------------------------------------------------
// Multi-head attention for MI455X (gfx1250), wave32 + WMMA 16x16x32 f16,
// async global->LDS staging (ASYNCcnt) + double-buffered K loops.
//   B=2, S=2048, D=1024, H=16, d_k=64.  Outputs: [output fp32][attn fp32].
// ---------------------------------------------------------------------------

typedef __attribute__((ext_vector_type(16))) _Float16 v16h;
typedef __attribute__((ext_vector_type(8)))  _Float16 v8h;
typedef __attribute__((ext_vector_type(4)))  _Float16 v4h;
typedef __attribute__((ext_vector_type(8)))  float    v8f;
typedef __attribute__((ext_vector_type(4)))  float    v4f;
typedef __attribute__((ext_vector_type(4)))  int      i32x4;

constexpr int D_MODEL   = 1024;
constexpr int NUM_HEADS = 16;
constexpr int D_K       = 64;
constexpr int SEQ       = 2048;
constexpr int BATCH     = 2;
constexpr int M_ROWS    = BATCH * SEQ;          // 4096

// ---------------------------------------------------------------------------
// CDNA5 async global->LDS copy (ASYNCcnt path), guarded with a sync fallback.
// Probe result: builtin signature is (i32x4 addrspace(1)*, i32x4 addrspace(3)*,
// imm offset, imm cpol).
// ---------------------------------------------------------------------------

#if defined(__gfx1250__) &&                                              \
    __has_builtin(__builtin_amdgcn_global_load_async_to_lds_b128) &&     \
    __has_builtin(__builtin_amdgcn_s_wait_asynccnt)
#define HAVE_ASYNC_LDS 1
#else
#define HAVE_ASYNC_LDS 0
#endif

__device__ inline void async_copy16(const _Float16* g, _Float16* l) {
#if HAVE_ASYNC_LDS
  __builtin_amdgcn_global_load_async_to_lds_b128(
      (__attribute__((address_space(1))) i32x4*)g,
      (__attribute__((address_space(3))) i32x4*)l, /*offset=*/0, /*cpol=*/0);
#else
  *(v8h*)l = *(const v8h*)g;
#endif
}

#if HAVE_ASYNC_LDS
#define WAIT_ASYNC(n) __builtin_amdgcn_s_wait_asynccnt(n)
#else
#define WAIT_ASYNC(n) do {} while (0)
#endif

// ---------------------------------------------------------------------------
// WMMA helpers (CDNA5 VGPR layouts, CDNA5 ISA §7.12.2)
// ---------------------------------------------------------------------------

__device__ inline void zero_acc(v8f& a) {
#pragma unroll
  for (int e = 0; e < 8; ++e) a[e] = 0.0f;
}

// A fragment 16x32 (MxK), f16, from LDS row-major tile (row stride ldk halves).
// Lane L: M = L%16 ; halves 0..7 -> K = 8g..8g+7 ; halves 8..15 -> K = 16+8g..+7
__device__ inline v16h load_a_frag(const _Float16* base, int ldk) {
  const int lane = threadIdx.x & 31;
  const int r = lane & 15, g = lane >> 4;
  const _Float16* p = base + r * ldk + 8 * g;
  v16h a;
  *((v8h*)&a)     = *(const v8h*)(p);        // ds_load_b128
  *((v8h*)&a + 1) = *(const v8h*)(p + 16);   // ds_load_b128
  return a;
}

// B fragment 32x16 (KxN), f16, from LDS tile stored TRANSPOSED: element (k,n)
// at bt[n*ldk + k].  Lane L: N = L%16 ; halves 0..7 -> K = 8g..8g+7 ; halves 8..15 -> K = 16+8g..+7
__device__ inline v16h load_b_frag(const _Float16* bt, int ldk) {
  const int lane = threadIdx.x & 31;
  const int n = lane & 15, g = lane >> 4;
  const _Float16* p = bt + n * ldk + 8 * g;
  v16h b;
  *((v8h*)&b)     = *(const v8h*)(p);
  *((v8h*)&b + 1) = *(const v8h*)(p + 16);
  return b;
}

__device__ inline v8f wmma_f16(v16h a, v16h b, v8f c) {
  return __builtin_amdgcn_wmma_f32_16x16x32_f16(
      false, a, false, b, (short)0, c, false, false);
}

// ---------------------------------------------------------------------------
// Kernel 0: fp32 -> f16 convert (vectorized, coalesced).  n = 4,194,304.
// grid 1024 x 256 threads, 16 elements/thread.
// ---------------------------------------------------------------------------

__global__ __launch_bounds__(256) void f32_to_f16_kernel(
    const float* __restrict__ in, _Float16* __restrict__ out) {
  const size_t blk = (size_t)blockIdx.x * 4096;
#pragma unroll
  for (int i = 0; i < 4; ++i) {
    const size_t off = blk + (size_t)i * 1024 + (size_t)threadIdx.x * 4;
    v4f x = *(const v4f*)(in + off);
    v4h h;
#pragma unroll
    for (int j = 0; j < 4; ++j) h[j] = (_Float16)x[j];
    *(v4h*)(out + off) = h;
  }
}

// ---------------------------------------------------------------------------
// Kernel 1/5: y = X @ W + b   (M=4096, N=K=1024), X is f16, W fp32.
// OUT_MODE 0: f16 [B,H,S,d_k] head layout -> ws.  OUT_MODE 1: fp32 [M,N].
// Block 256 thr (8 waves), tile 128x128, K-step 32.
// A tile: double-buffered LDS via async global->LDS copies.
// B tile: register-pipelined (load t+1 during compute of t), transposed store.
// ---------------------------------------------------------------------------

template <int OUT_MODE>
__global__ __launch_bounds__(256) void gemm1024_kernel(
    const _Float16* __restrict__ A, const float* __restrict__ Bw,
    const float* __restrict__ bias, void* __restrict__ Out) {
  constexpr int BK = 32, LDK = BK + 8;     // pad keeps rows 16B aligned
  __shared__ alignas(16) _Float16 As[2][128 * LDK];
  __shared__ alignas(16) _Float16 Bt[128 * LDK];

  const int tid  = threadIdx.x;
  const int lane = tid & 31;
  const int wave = tid >> 5;
  const int wrow = (wave >> 1) * 32;       // 0,32,64,96
  const int wcol = (wave & 1) * 64;        // 0,64
  const int mBase = blockIdx.y * 128;
  const int nBase = blockIdx.x * 128;

  const _Float16* Ablk = A + (size_t)mBase * D_MODEL;

  // A tile 128x32 f16 = 512 x (8 halves); 2 async b128 copies per thread.
  auto issueA = [&](int kt, _Float16* dst) {
#pragma unroll
    for (int i = 0; i < 2; ++i) {
      const int c = tid + 256 * i;
      const int row = c >> 2, c8 = (c & 3) * 8;
      async_copy16(Ablk + (size_t)row * D_MODEL + kt + c8,
                   dst + row * LDK + c8);
    }
  };
  // B tile 32x128 fp32 = 1024 float4; 4 per thread, held in registers.
  auto loadB = [&](int kt, v4f* br) {
#pragma unroll
    for (int i = 0; i < 4; ++i) {
      const int f = tid + 256 * i;
      const int kr = f >> 5, c4 = (f & 31) * 4;
      br[i] = *(const v4f*)(Bw + (size_t)(kt + kr) * D_MODEL + nBase + c4);
    }
  };
  auto storeB = [&](const v4f* br) {
#pragma unroll
    for (int i = 0; i < 4; ++i) {
      const int f = tid + 256 * i;
      const int kr = f >> 5, c4 = (f & 31) * 4;
#pragma unroll
      for (int j = 0; j < 4; ++j)
        Bt[(c4 + j) * LDK + kr] = (_Float16)br[i][j];
    }
  };

  v8f acc[2][4];
#pragma unroll
  for (int i = 0; i < 2; ++i)
#pragma unroll
    for (int j = 0; j < 4; ++j) zero_acc(acc[i][j]);

  v4f breg[4];
  issueA(0, As[0]);
  loadB(0, breg);

  constexpr int NT = D_MODEL / BK;         // 32 K-tiles
  for (int t = 0; t < NT; ++t) {
    _Float16* curA = As[t & 1];
    _Float16* nxtA = As[(t & 1) ^ 1];

    storeB(breg);                          // Bt <- tile t (prev compute done)
    if (t + 1 < NT) {
      issueA((t + 1) * BK, nxtA);          // prefetch A tile t+1 (async)
      WAIT_ASYNC(2);                       // only tile t+1's 2 copies in flight
    } else {
      WAIT_ASYNC(0);
    }
    __syncthreads();                       // all waves' tile-t data visible
    if (t + 1 < NT) loadB((t + 1) * BK, breg);  // overlap with WMMA below

    v16h af[2], bf[4];
#pragma unroll
    for (int i = 0; i < 2; ++i)
      af[i] = load_a_frag(&curA[(wrow + 16 * i) * LDK], LDK);
#pragma unroll
    for (int j = 0; j < 4; ++j)
      bf[j] = load_b_frag(&Bt[(wcol + 16 * j) * LDK], LDK);
#pragma unroll
    for (int i = 0; i < 2; ++i)
#pragma unroll
      for (int j = 0; j < 4; ++j) acc[i][j] = wmma_f16(af[i], bf[j], acc[i][j]);
    __syncthreads();
  }

  // Epilogue: C/D layout is (M = e + 8*(lane/16), N = lane%16)
  const int g = lane >> 4, nlo = lane & 15;
#pragma unroll
  for (int i = 0; i < 2; ++i) {
#pragma unroll
    for (int j = 0; j < 4; ++j) {
      const int col = nBase + wcol + 16 * j + nlo;
      const float bv = bias[col];
#pragma unroll
      for (int e = 0; e < 8; ++e) {
        const int row = mBase + wrow + 16 * i + e + 8 * g;
        const float val = acc[i][j][e] + bv;
        if (OUT_MODE == 0) {
          const int b = row >> 11, s = row & (SEQ - 1);
          const int h = col >> 6,  d = col & (D_K - 1);
          ((_Float16*)Out)[(((size_t)b * NUM_HEADS + h) * SEQ + s) * D_K + d] =
              (_Float16)val;
        } else {
          ((float*)Out)[(size_t)row * D_MODEL + col] = val;
        }
      }
    }
  }
}

// ---------------------------------------------------------------------------
// Kernel 2/5: raw scores  S = (qh @ kh^T) / sqrt(d_k)  -> attn region (fp32)
// grid: (key_tile 16, q_tile 16, b*H 32), block 256, tile 128x128, K=d_k=64.
// Q/K tiles staged with async global->LDS copies (straight f16 copy; kh's
// [key][dim] head layout IS the transposed-B layout WMMA needs).
// ---------------------------------------------------------------------------

__global__ __launch_bounds__(256) void scores_kernel(
    const _Float16* __restrict__ qh, const _Float16* __restrict__ kh,
    float* __restrict__ attn) {
  constexpr int LDK = D_K + 8;             // 72 halves, rows 16B aligned
  __shared__ alignas(16) _Float16 Qs[128 * LDK];
  __shared__ alignas(16) _Float16 Ks[128 * LDK];

  const int tid  = threadIdx.x;
  const int lane = tid & 31;
  const int wave = tid >> 5;
  const int wrow = (wave >> 1) * 32;
  const int wcol = (wave & 1) * 64;
  const int bh = blockIdx.z;

  const _Float16* qbase =
      qh + (size_t)bh * SEQ * D_K + (size_t)blockIdx.y * 128 * D_K;
  const _Float16* kbase =
      kh + (size_t)bh * SEQ * D_K + (size_t)blockIdx.x * 128 * D_K;

  // 128 rows x 64 halves each = 1024 b128 chunks per matrix; 4+4 per thread
#pragma unroll
  for (int i = 0; i < 4; ++i) {
    const int c = tid + 256 * i;
    const int row = c >> 3, c8 = (c & 7) * 8;
    async_copy16(qbase + (size_t)row * D_K + c8, &Qs[row * LDK + c8]);
    async_copy16(kbase + (size_t)row * D_K + c8, &Ks[row * LDK + c8]);
  }
  WAIT_ASYNC(0);
  __syncthreads();

  v8f acc[2][4];
#pragma unroll
  for (int i = 0; i < 2; ++i)
#pragma unroll
    for (int j = 0; j < 4; ++j) zero_acc(acc[i][j]);

#pragma unroll
  for (int ks = 0; ks < D_K; ks += 32) {
    v16h af[2], bf[4];
#pragma unroll
    for (int i = 0; i < 2; ++i)
      af[i] = load_a_frag(&Qs[(wrow + 16 * i) * LDK + ks], LDK);
#pragma unroll
    for (int j = 0; j < 4; ++j)
      bf[j] = load_b_frag(&Ks[(wcol + 16 * j) * LDK + ks], LDK);
#pragma unroll
    for (int i = 0; i < 2; ++i)
#pragma unroll
      for (int j = 0; j < 4; ++j) acc[i][j] = wmma_f16(af[i], bf[j], acc[i][j]);
  }

  const float scale = 0.125f;              // 1/sqrt(64)
  const int g = lane >> 4, nlo = lane & 15;
  float* outbh = attn + (size_t)bh * SEQ * SEQ;
#pragma unroll
  for (int i = 0; i < 2; ++i)
#pragma unroll
    for (int j = 0; j < 4; ++j)
#pragma unroll
      for (int e = 0; e < 8; ++e) {
        const int row = blockIdx.y * 128 + wrow + 16 * i + e + 8 * g;
        const int col = blockIdx.x * 128 + wcol + 16 * j + nlo;
        outbh[(size_t)row * SEQ + col] = acc[i][j][e] * scale;
      }
}

// ---------------------------------------------------------------------------
// Kernel 3/5: in-place softmax over each attn row.  One block per row,
// 256 threads x 8 elements.  Two block reductions (shfl_xor + LDS).
// ---------------------------------------------------------------------------

__global__ __launch_bounds__(256) void softmax_kernel(float* __restrict__ attn) {
  float* p = attn + (size_t)blockIdx.x * SEQ;
  const int tid = threadIdx.x;
  const int lane = tid & 31, wave = tid >> 5;

  float v[8];
  float m = -3.0e38f;
#pragma unroll
  for (int i = 0; i < 8; ++i) {
    v[i] = p[tid + 256 * i];
    m = fmaxf(m, v[i]);
  }
#pragma unroll
  for (int off = 16; off > 0; off >>= 1) m = fmaxf(m, __shfl_xor(m, off, 32));

  __shared__ float red[8];
  if (lane == 0) red[wave] = m;
  __syncthreads();
  float bm = red[0];
#pragma unroll
  for (int w = 1; w < 8; ++w) bm = fmaxf(bm, red[w]);

  float s = 0.0f;
#pragma unroll
  for (int i = 0; i < 8; ++i) {
    v[i] = __expf(v[i] - bm);
    s += v[i];
  }
#pragma unroll
  for (int off = 16; off > 0; off >>= 1) s += __shfl_xor(s, off, 32);
  __syncthreads();
  if (lane == 0) red[wave] = s;
  __syncthreads();
  float bs = 0.0f;
#pragma unroll
  for (int w = 0; w < 8; ++w) bs += red[w];

  const float inv = 1.0f / bs;
#pragma unroll
  for (int i = 0; i < 8; ++i) p[tid + 256 * i] = v[i] * inv;
}

// ---------------------------------------------------------------------------
// Kernel 4/5: ctx = attn @ vh per (b,h).  M=2048, N=64, K=2048.
// grid: (m_tile 16, b*H 32), block 256, tile 128x64, K-step 32.
// Register-pipelined: global loads for tile t+1 issued before computing t.
// attn loaded non-temporally (each element read exactly once).
// ---------------------------------------------------------------------------

__global__ __launch_bounds__(256) void ctx_kernel(
    const float* __restrict__ attn, const _Float16* __restrict__ vh,
    _Float16* __restrict__ ctx) {
  constexpr int LDK = 40;
  __shared__ alignas(16) _Float16 As[128 * LDK];
  __shared__ alignas(16) _Float16 Bt[64 * LDK];

  const int tid  = threadIdx.x;
  const int lane = tid & 31;
  const int wave = tid >> 5;
  const int wrow = (wave >> 1) * 32;
  const int wcol = (wave & 1) * 32;
  const int bh = blockIdx.y;
  const int b = bh >> 4, h = bh & 15;

  const float* abase =
      attn + (size_t)bh * SEQ * SEQ + (size_t)blockIdx.x * 128 * SEQ;
  const _Float16* vbase = vh + (size_t)bh * SEQ * D_K;

  const int arow = tid >> 3, ac4 = (tid & 7) * 4;   // fixed per-thread slots
  const int vkr  = tid >> 3, vc8 = (tid & 7) * 8;

  auto loadA = [&](int kt, v4f* ar) {
#pragma unroll
    for (int i = 0; i < 4; ++i)
      ar[i] = __builtin_nontemporal_load(
          (const v4f*)(abase + (size_t)(arow + 32 * i) * SEQ + kt + ac4));
  };
  auto loadV = [&](int kt, v8h& vr) {
    vr = *(const v8h*)(vbase + (size_t)(kt + vkr) * D_K + vc8);
  };

  v8f acc[2][2];
#pragma unroll
  for (int i = 0; i < 2; ++i)
#pragma unroll
    for (int j = 0; j < 2; ++j) zero_acc(acc[i][j]);

  v4f areg[4];
  v8h vreg;
  loadA(0, areg);
  loadV(0, vreg);

  for (int kt = 0; kt < SEQ; kt += 32) {
    // stage tile t from registers
#pragma unroll
    for (int i = 0; i < 4; ++i) {
      v4h hv;
#pragma unroll
      for (int j = 0; j < 4; ++j) hv[j] = (_Float16)areg[i][j];
      *(v4h*)&As[(arow + 32 * i) * LDK + ac4] = hv;
    }
#pragma unroll
    for (int j = 0; j < 8; ++j) Bt[(vc8 + j) * LDK + vkr] = vreg[j];
    __syncthreads();

    if (kt + 32 < SEQ) {                   // prefetch tile t+1 during compute
      loadA(kt + 32, areg);
      loadV(kt + 32, vreg);
    }

    v16h af[2], bf[2];
#pragma unroll
    for (int i = 0; i < 2; ++i)
      af[i] = load_a_frag(&As[(wrow + 16 * i) * LDK], LDK);
#pragma unroll
    for (int j = 0; j < 2; ++j)
      bf[j] = load_b_frag(&Bt[(wcol + 16 * j) * LDK], LDK);
#pragma unroll
    for (int i = 0; i < 2; ++i)
#pragma unroll
      for (int j = 0; j < 2; ++j) acc[i][j] = wmma_f16(af[i], bf[j], acc[i][j]);
    __syncthreads();
  }

  const int g = lane >> 4, nlo = lane & 15;
#pragma unroll
  for (int i = 0; i < 2; ++i)
#pragma unroll
    for (int j = 0; j < 2; ++j)
#pragma unroll
      for (int e = 0; e < 8; ++e) {
        const int row = blockIdx.x * 128 + wrow + 16 * i + e + 8 * g;  // s
        const int col = wcol + 16 * j + nlo;                           // d
        ctx[((size_t)b * SEQ + row) * D_MODEL + h * D_K + col] =
            (_Float16)acc[i][j][e];
      }
}

// ---------------------------------------------------------------------------
// Launch.  Workspace (f16): qh | kh | vh | ctx | X (reused convert buffer)
//          = 5 x 8 MB = 40 MB.   d_out = [output 4.19M fp32][attn 134.2M fp32]
// ---------------------------------------------------------------------------

extern "C" void kernel_launch(void* const* d_in, const int* in_sizes, int n_in,
                              void* d_out, int out_size, void* d_ws,
                              size_t ws_size, hipStream_t stream) {
  const float* q  = (const float*)d_in[0];
  const float* k  = (const float*)d_in[1];
  const float* v  = (const float*)d_in[2];
  const float* wq = (const float*)d_in[3];
  const float* bq = (const float*)d_in[4];
  const float* wk = (const float*)d_in[5];
  const float* bk = (const float*)d_in[6];
  const float* wv = (const float*)d_in[7];
  const float* bv = (const float*)d_in[8];
  const float* wo = (const float*)d_in[9];
  const float* bo = (const float*)d_in[10];

  float* out  = (float*)d_out;
  float* attn = out + (size_t)M_ROWS * D_MODEL;

  const size_t HBUF = (size_t)M_ROWS * D_MODEL;        // halves per buffer
  _Float16* qh  = (_Float16*)d_ws;
  _Float16* kh  = qh + HBUF;
  _Float16* vh  = kh + HBUF;
  _Float16* ctx = vh + HBUF;
  _Float16* X   = ctx + HBUF;                          // reused f16 input

  const dim3 blk(256);
  const dim3 gConv(M_ROWS * D_MODEL / 4096);           // 1024
  const dim3 gProj(D_MODEL / 128, M_ROWS / 128);       // (8, 32)

  // 1) convert + project Q/K/V -> f16 head layout in workspace
  f32_to_f16_kernel<<<gConv, blk, 0, stream>>>(q, X);
  gemm1024_kernel<0><<<gProj, blk, 0, stream>>>(X, wq, bq, (void*)qh);
  f32_to_f16_kernel<<<gConv, blk, 0, stream>>>(k, X);
  gemm1024_kernel<0><<<gProj, blk, 0, stream>>>(X, wk, bk, (void*)kh);
  f32_to_f16_kernel<<<gConv, blk, 0, stream>>>(v, X);
  gemm1024_kernel<0><<<gProj, blk, 0, stream>>>(X, wv, bv, (void*)vh);

  // 2) raw scores -> attn region of d_out
  scores_kernel<<<dim3(SEQ / 128, SEQ / 128, BATCH * NUM_HEADS), blk, 0,
                  stream>>>(qh, kh, attn);

  // 3) softmax in place (one block per row)
  softmax_kernel<<<dim3(BATCH * NUM_HEADS * SEQ), blk, 0, stream>>>(attn);

  // 4) ctx = attn @ vh -> f16 [B,S,D] in workspace
  ctx_kernel<<<dim3(SEQ / 128, BATCH * NUM_HEADS), blk, 0, stream>>>(attn, vh,
                                                                     ctx);

  // 5) output projection -> fp32 output region of d_out
  gemm1024_kernel<1><<<gProj, blk, 0, stream>>>(ctx, wo, bo, (void*)out);
}
